// MHSA_8770323218690
// MI455X (gfx1250) — matrix operand order repaired and checked by reading, not compiled, not run on hardware
//
#include <hip/hip_runtime.h>

// MHSA for B=2, S=2048, D=1024, H=16, hd=64 on gfx1250 (MI455X).
//   1) qkv_proj : x[4096,1024] x Wqkv^T -> q/k/v (f16), v stored transposed
//   2) attn     : flash attention per (b,h,qtile), online softmax, P bounced via LDS
//   3) out_proj : ctx[4096,1024](f16) x Wo^T -> y (f32) + bias; A tile staged with
//                 GLOBAL_LOAD_ASYNC_TO_LDS_B128 + s_wait_asynccnt (CDNA5 async path)

typedef __attribute__((ext_vector_type(16))) _Float16 v16h;
typedef __attribute__((ext_vector_type(8)))  _Float16 v8h;
typedef __attribute__((ext_vector_type(8)))  float    v8f;

#define LDT 40   // GEMM LDS tile stride in halves: 32 + 8 pad (80B rows -> 16B aligned, bank-spread)
#define LDP 80   // attention P-tile stride in halves: 64 + 16 pad (160B rows -> 16B aligned)

__device__ __forceinline__ v8f zero8() {
    v8f z;
#pragma unroll
    for (int i = 0; i < 8; ++i) z[i] = 0.0f;
    return z;
}

// A-matrix fragment (16x32 f16): lane r=lane&15 is row r; halves = K[klo..klo+7], K[klo+16..klo+23]
__device__ __forceinline__ v16h frag_a(const _Float16* rowp, int klo) {
    const _Float16* p = rowp + klo;
    v8h lo  = *(const v8h*)(p);
    v8h hi8 = *(const v8h*)(p + 16);
    v16h f;
#pragma unroll
    for (int i = 0; i < 8; ++i) { f[i] = lo[i]; f[i + 8] = hi8[i]; }
    return f;
}

// B-matrix fragment (32x16 f16): lane n=lane&15 is column n; halves = K[koff16/2..koff16/2+7], K[koff16/2+16..koff16/2+23]
__device__ __forceinline__ v16h frag_b(const _Float16* colp, int koff16) {
    const _Float16* p = colp + (koff16 >> 1);
    v8h lo  = *(const v8h*)(p);
    v8h hi8 = *(const v8h*)(p + 16);
    v16h f;
#pragma unroll
    for (int i = 0; i < 8; ++i) { f[i] = lo[i]; f[i + 8] = hi8[i]; }
    return f;
}

__device__ __forceinline__ v8f mma(v16h a, v16h b, v8f c) {
    return __builtin_amdgcn_wmma_f32_16x16x32_f16(false, a, false, b, (short)0, c, false, false);
}

// --- CDNA5 async copy: 16B global -> LDS per lane, tracked on ASYNCcnt ---
__device__ __forceinline__ void async_ld128(unsigned int lds_off, const void* gptr) {
    asm volatile("global_load_async_to_lds_b128 %0, %1, off"
                 :: "v"(lds_off), "v"((unsigned long long)(uintptr_t)gptr)
                 : "memory");
}
__device__ __forceinline__ void wait_async0() {
    asm volatile("s_wait_asynccnt 0x0" ::: "memory");
}

// Stage a ROWSx32 fp32 tile into LDS as f16 (row stride LDT halves). 128 threads.
template<int ROWS>
__device__ __forceinline__ void stage_f32(const float* __restrict__ g, int ldg,
                                          _Float16* __restrict__ smem, int tid) {
#pragma unroll
    for (int it = 0; it < ROWS / 16; ++it) {
        int q = tid + it * 128;          // ROWS x 8 float4 chunks
        int r = q >> 3;
        int c = (q & 7) * 4;
        float4 x = *(const float4*)(g + r * ldg + c);
        _Float16* d = smem + r * LDT + c;
        d[0] = (_Float16)x.x; d[1] = (_Float16)x.y;
        d[2] = (_Float16)x.z; d[3] = (_Float16)x.w;
    }
}

// Stage a 64x32 f16 tile into LDS via async b128 copies (no VGPR bounce). 128 threads.
__device__ __forceinline__ void stage_f16_async(const _Float16* __restrict__ g, int ldg,
                                                _Float16* __restrict__ smem, int tid) {
#pragma unroll
    for (int it = 0; it < 2; ++it) {
        int q = tid + it * 128;          // 0..255 : 64 rows x 4 b128 chunks
        int r = q >> 2;
        int c = (q & 3) * 8;
        async_ld128((unsigned int)(uintptr_t)(smem + r * LDT + c), g + r * ldg + c);
    }
}

// ---------------------------------------------------------------------------
// Kernel 1: QKV projection. qkv[m,e] = sum_k x[m,k]*Wqkv[e,k] + bqkv[e]
// Block = 128 thr (4 waves), tile 64x128 (wave tile 32x64), K-steps of 32.
// Scatter: e<1024 -> q[B,H,S,hd]; e<2048 -> k[B,H,S,hd]; else -> vT[B,H,hd,S]
// ---------------------------------------------------------------------------
__global__ __launch_bounds__(128)
void mhsa_qkv_proj(const float* __restrict__ x, const float* __restrict__ Wqkv,
                   const float* __restrict__ bqkv,
                   _Float16* __restrict__ qbuf, _Float16* __restrict__ kbuf,
                   _Float16* __restrict__ vtbuf) {
    __shared__ __align__(16) _Float16 As[64 * LDT];
    __shared__ __align__(16) _Float16 Bs[128 * LDT];
    const int tid = threadIdx.x, lane = tid & 31, wave = tid >> 5;
    const int mBlk = blockIdx.x * 64, nBlk = blockIdx.y * 128;
    const int wm = (wave >> 1) * 32, wn = (wave & 1) * 64;
    const int r = lane & 15, hi = lane >> 4;
    const int klo = hi * 8, koff16 = hi * 16;

    v8f acc[2][4];
#pragma unroll
    for (int i = 0; i < 2; ++i)
#pragma unroll
        for (int j = 0; j < 4; ++j) acc[i][j] = zero8();

    for (int k0 = 0; k0 < 1024; k0 += 32) {
        __syncthreads();
        stage_f32<64>(x + mBlk * 1024 + k0, 1024, As, tid);
        stage_f32<128>(Wqkv + nBlk * 1024 + k0, 1024, Bs, tid);
        __syncthreads();
        v16h a0 = frag_a(As + (wm + r) * LDT, klo);
        v16h a1 = frag_a(As + (wm + 16 + r) * LDT, klo);
#pragma unroll
        for (int tn = 0; tn < 4; ++tn) {
            v16h bf = frag_b(Bs + (wn + tn * 16 + r) * LDT, koff16);
            acc[0][tn] = mma(a0, bf, acc[0][tn]);
            acc[1][tn] = mma(a1, bf, acc[1][tn]);
        }
    }

#pragma unroll
    for (int tm = 0; tm < 2; ++tm)
#pragma unroll
        for (int tn = 0; tn < 4; ++tn)
#pragma unroll
            for (int v = 0; v < 8; ++v) {
                int m = mBlk + wm + tm * 16 + v + 8 * hi;   // 0..4095
                int e = nBlk + wn + tn * 16 + r;            // 0..3071
                float val = acc[tm][tn][v] + bqkv[e];
                int b = m >> 11, s = m & 2047;
                int sec = e >> 10, ei = e & 1023;
                int h = ei >> 6, d = ei & 63;
                int bh = b * 16 + h;
                _Float16 hv = (_Float16)val;
                if (sec == 0)      qbuf[(bh * 2048 + s) * 64 + d] = hv;
                else if (sec == 1) kbuf[(bh * 2048 + s) * 64 + d] = hv;
                else               vtbuf[(bh * 64 + d) * 2048 + s] = hv;
            }
}

// ---------------------------------------------------------------------------
// Kernel 2: flash attention. Block = 4 waves; wave owns 16 q-rows (64/block).
// ---------------------------------------------------------------------------
__global__ __launch_bounds__(128)
void mhsa_flash_attn(const _Float16* __restrict__ qbuf, const _Float16* __restrict__ kbuf,
                     const _Float16* __restrict__ vtbuf, _Float16* __restrict__ ctx) {
    __shared__ __align__(16) _Float16 Pl[4][16 * LDP];
    const int tid = threadIdx.x, lane = tid & 31, wave = tid >> 5;
    const int qt = blockIdx.x, h = blockIdx.y, b = blockIdx.z;
    const int bh = b * 16 + h;
    const int qBase = qt * 64 + wave * 16;
    const int r = lane & 15, hi = lane >> 4;
    const int klo = hi * 8, koff16 = hi * 16;

    const _Float16* qp = qbuf + (bh * 2048 + qBase) * 64;
    const _Float16* kp = kbuf + bh * 2048 * 64;
    const _Float16* vp = vtbuf + bh * 64 * 2048;

    v16h qa[2];
#pragma unroll
    for (int c = 0; c < 2; ++c) qa[c] = frag_a(qp + r * 64 + c * 32, klo);

    float mrow[8], lrow[8];
    v8f o[4];
#pragma unroll
    for (int v = 0; v < 8; ++v) { mrow[v] = -1.0e30f; lrow[v] = 0.0f; }
#pragma unroll
    for (int nt = 0; nt < 4; ++nt) o[nt] = zero8();

    _Float16* pw = &Pl[wave][0];

    for (int kb = 0; kb < 2048; kb += 64) {
        if (kb + 64 < 2048) {   // uniform branch: prefetch next K/V block into caches
            __builtin_prefetch(kp + (kb + 64 + r) * 64, 0, 1);
            __builtin_prefetch(vp + r * 2048 + kb + 64, 0, 1);
        }
        // --- scores: S[16 x 64] = Q(16x64) * K^T ---
        v8f sc[4];
#pragma unroll
        for (int nt = 0; nt < 4; ++nt) {
            sc[nt] = zero8();
            const _Float16* kcol = kp + (kb + nt * 16 + r) * 64;  // lane's key row
            sc[nt] = mma(qa[0], frag_b(kcol, koff16), sc[nt]);
            sc[nt] = mma(qa[1], frag_b(kcol + 32, koff16), sc[nt]);
        }
        // --- online softmax (scale = 1/sqrt(64) = 0.125) ---
#pragma unroll
        for (int v = 0; v < 8; ++v) {
            float x0 = fmaxf(fmaxf(sc[0][v], sc[1][v]), fmaxf(sc[2][v], sc[3][v])) * 0.125f;
#pragma unroll
            for (int msk = 1; msk < 16; msk <<= 1) x0 = fmaxf(x0, __shfl_xor(x0, msk, 32));
            float mnew = fmaxf(mrow[v], x0);
            float corr = __expf(mrow[v] - mnew);
            mrow[v] = mnew;
            float psum = 0.0f;
#pragma unroll
            for (int nt = 0; nt < 4; ++nt) {
                float p = __expf(sc[nt][v] * 0.125f - mnew);
                sc[nt][v] = p;
                psum += p;
            }
#pragma unroll
            for (int msk = 1; msk < 16; msk <<= 1) psum += __shfl_xor(psum, msk, 32);
            lrow[v] = lrow[v] * corr + psum;
#pragma unroll
            for (int nt = 0; nt < 4; ++nt) o[nt][v] *= corr;
        }
        // --- P: C-layout regs -> LDS -> A-layout frags (wave-private tile) ---
#pragma unroll
        for (int nt = 0; nt < 4; ++nt)
#pragma unroll
            for (int v = 0; v < 8; ++v)
                pw[(v + 8 * hi) * LDP + nt * 16 + r] = (_Float16)sc[nt][v];
        v16h pa0 = frag_a(pw + r * LDP, klo);
        v16h pa1 = frag_a(pw + r * LDP + 32, klo);
        // --- O += P(16x64) * V(64x64), V transposed so K-dim is contiguous ---
#pragma unroll
        for (int nt = 0; nt < 4; ++nt) {
            const _Float16* vcol = vp + (nt * 16 + r) * 2048 + kb;  // lane's hd column
            o[nt] = mma(pa0, frag_b(vcol, koff16), o[nt]);
            o[nt] = mma(pa1, frag_b(vcol + 32, koff16), o[nt]);
        }
    }

    _Float16* cp = ctx + (b * 2048 + qBase) * 1024 + h * 64;
#pragma unroll
    for (int nt = 0; nt < 4; ++nt)
#pragma unroll
        for (int v = 0; v < 8; ++v) {
            int m = v + 8 * hi;
            cp[m * 1024 + nt * 16 + r] = (_Float16)(o[nt][v] / lrow[v]);
        }
}

// ---------------------------------------------------------------------------
// Kernel 3: output projection. y[m,n] = sum_k ctx[m,k]*Wo[n,k] + bo[n]  (f32 out)
// Block tile 64x128 (wave tile 32x64); ctx tile staged via async-to-LDS.
// ---------------------------------------------------------------------------
__global__ __launch_bounds__(128)
void mhsa_out_proj(const _Float16* __restrict__ ctx, const float* __restrict__ Wo,
                   const float* __restrict__ bo, float* __restrict__ y) {
    __shared__ __align__(16) _Float16 As[64 * LDT];
    __shared__ __align__(16) _Float16 Bs[128 * LDT];
    const int tid = threadIdx.x, lane = tid & 31, wave = tid >> 5;
    const int mBlk = blockIdx.x * 64, nBlk = blockIdx.y * 128;
    const int wm = (wave >> 1) * 32, wn = (wave & 1) * 64;
    const int r = lane & 15, hi = lane >> 4;
    const int klo = hi * 8, koff16 = hi * 16;

    v8f acc[2][4];
#pragma unroll
    for (int i = 0; i < 2; ++i)
#pragma unroll
        for (int j = 0; j < 4; ++j) acc[i][j] = zero8();

    for (int k0 = 0; k0 < 1024; k0 += 32) {
        __syncthreads();
        stage_f16_async(ctx + mBlk * 1024 + k0, 1024, As, tid);   // ASYNCcnt path
        stage_f32<128>(Wo + nBlk * 1024 + k0, 1024, Bs, tid);     // convert path
        wait_async0();
        __syncthreads();
        v16h a0 = frag_a(As + (wm + r) * LDT, klo);
        v16h a1 = frag_a(As + (wm + 16 + r) * LDT, klo);
#pragma unroll
        for (int tn = 0; tn < 4; ++tn) {
            v16h bf = frag_b(Bs + (wn + tn * 16 + r) * LDT, koff16);
            acc[0][tn] = mma(a0, bf, acc[0][tn]);
            acc[1][tn] = mma(a1, bf, acc[1][tn]);
        }
    }

#pragma unroll
    for (int tm = 0; tm < 2; ++tm)
#pragma unroll
        for (int tn = 0; tn < 4; ++tn)
#pragma unroll
            for (int v = 0; v < 8; ++v) {
                int m = mBlk + wm + tm * 16 + v + 8 * hi;
                int n = nBlk + wn + tn * 16 + r;
                y[m * 1024 + n] = acc[tm][tn][v] + bo[n];
            }
}

extern "C" void kernel_launch(void* const* d_in, const int* in_sizes, int n_in,
                              void* d_out, int out_size, void* d_ws, size_t ws_size,
                              hipStream_t stream) {
    (void)in_sizes; (void)n_in; (void)out_size; (void)ws_size;
    const float* x    = (const float*)d_in[0];
    const float* Wqkv = (const float*)d_in[1];
    const float* bqkv = (const float*)d_in[2];
    const float* Wo   = (const float*)d_in[3];
    const float* bo   = (const float*)d_in[4];
    float* y = (float*)d_out;

    // f16 workspace: q | k | vT | ctx, each B*H*S*hd = 4M halves (8 MB); 32 MB total.
    _Float16* ws    = (_Float16*)d_ws;
    _Float16* qbuf  = ws;
    _Float16* kbuf  = ws + (size_t)4 * 1024 * 1024;
    _Float16* vtbuf = ws + (size_t)8 * 1024 * 1024;
    _Float16* ctx   = ws + (size_t)12 * 1024 * 1024;

    mhsa_qkv_proj<<<dim3(64, 24), 128, 0, stream>>>(x, Wqkv, bqkv, qbuf, kbuf, vtbuf);
    mhsa_flash_attn<<<dim3(32, 16, 2), 128, 0, stream>>>(qbuf, kbuf, vtbuf, ctx);
    mhsa_out_proj<<<dim3(64, 8), 128, 0, stream>>>(ctx, Wo, bo, y);
}
